// FourierShift_57698590655203
// MI455X (gfx1250) — compile-verified, matrix operand order checked by reading, not run on hardware
//
#include <hip/hip_runtime.h>
#include <hip/hip_bf16.h>

// ---------------------------------------------------------------------------
// FourierShift for MI455X (gfx1250, wave32, WMMA).
// One wave == one batch-time element for all estimator kernels.
// All GEMMs use v_wmma_f32_16x16x32_f16 (f16 in, f32 accumulate).
// B fragments are preloaded into registers in batches so the WMMA chain is not
// serialized on per-fragment L2 latency (loads clause together, wmmas issue
// back-to-back).
// ---------------------------------------------------------------------------

typedef __attribute__((ext_vector_type(8)))  _Float16 hvec8;
typedef __attribute__((ext_vector_type(16))) _Float16 hvec16;
typedef __attribute__((ext_vector_type(8)))  float    fvec8;

#define WMMA16(A, B, C) \
  __builtin_amdgcn_wmma_f32_16x16x32_f16(false, (A), false, (B), (short)0, (C), false, false)

#define PI_F 3.14159265358979f

// ---- packed f32 weight block offsets (floats, per estimator) ---------------
#define OFF_FC1   0     // 16x4
#define OFF_FC2   64    // 16x3
#define OFF_FC3   128   // 4x16
#define OFF_CONVW 192   // 8x4
#define OFF_CONVB 224   // 8
#define OFF_SE1W  232   // 4x8
#define OFF_SE1B  264   // 4
#define OFF_SE2W  268   // 8x4
#define OFF_SE2B  300   // 8
#define OFF_QB    308   // 64
#define OFF_KB    372   // 64
#define OFF_VB    436   // 64
#define OFF_OB    500   // 128
#define OFF_LAYER 640
#define LSTRIDE   1536
#define LOFF_LNG  0     // 128
#define LOFF_LNB  128   // 128
#define LOFF_F1W  256   // 64x8
#define LOFF_F1B  768   // 64
#define LOFF_BNG  832   // 32
#define LOFF_BNB  864   // 32
#define LOFF_PJB  896   // 128
#define LOFF_F2W  1024  // 8x32
#define LOFF_F2B  1280  // 8
#define ESTF32    5248
// f16 block (halfs, per estimator): qw 0, kw 8192, vw 16384, ow 24576, pj[L] 32768+16384L
#define ESTF16    81920

// ---------------------------------------------------------------------------
__device__ inline float sigmoidf_(float x) { return 1.0f / (1.0f + __expf(-x)); }
__device__ inline float softplusf_(float x) { return (x > 20.f) ? x : log1pf(__expf(x)); }
__device__ inline float geluf_(float x) { return 0.5f * x * (1.0f + erff(x * 0.70710678118f)); }

__device__ inline float wave_sum(float v) {
#pragma unroll
  for (int o = 16; o > 0; o >>= 1) v += __shfl_xor(v, o, 32);
  return v;
}

__device__ inline fvec8 zero8() {
  fvec8 z = {0.f, 0.f, 0.f, 0.f, 0.f, 0.f, 0.f, 0.f};
  return z;
}

// Load a 16x32 f16 fragment from a row-major matrix (rows are the "M"/"N"
// dimension, contiguous K).  Works for both A fragments (row-major A) and
// B fragments (row-major B^T, i.e. (out,in) weight layout).
// ISA layout: lane = m + 16*khalf ; VGPR0..3 -> K = khalf*8 .. +7,
//             VGPR4..7 -> K = 16 + khalf*8 .. +7.
__device__ inline hvec16 frag16(const _Float16* base, int ld) {
  const int lane = threadIdx.x & 31;
  const _Float16* r = base + (lane & 15) * ld + ((lane >> 4) << 3);
  hvec8 lo = *(const hvec8*)r;
  hvec8 hi = *(const hvec8*)(r + 16);
  hvec16 f;
#pragma unroll
  for (int i = 0; i < 8; ++i) { f[i] = lo[i]; f[i + 8] = hi[i]; }
  return f;
}

// ---------------------------------------------------------------------------
// K0: pack weights (f32 -> f16 GEMM blocks, f32 smalls), zero BN accumulators.
// Pointer table order = jax tree-flatten of params: proj, scale-leaves, shift-leaves.
// Per-estimator leaf order (sorted keys, gmlp list expanded in place):
//  0 conv_b 1 conv_w 2 fc1_w 3 fc2_w 4 fc3_w
//  5+10L: bn_b bn_g f1_b f1_w f2_b f2_w ln_b ln_g pj_b pj_w   (L = 0..2)
//  35 k_b 36 k_w 37 o_b 38 o_w 39 q_b 40 q_w 41 se1_b 42 se1_w 43 se2_b 44 se2_w 45 v_b 46 v_w
// ---------------------------------------------------------------------------
struct PtrTab { const float* p[94]; };

__device__ inline void copyf_(float* d, const float* s, int n, int t, int nt) {
  for (int i = t; i < n; i += nt) d[i] = s[i];
}
__device__ inline void copyh_(_Float16* d, const float* s, int n, int t, int nt) {
  for (int i = t; i < n; i += nt) d[i] = (_Float16)s[i];
}

__global__ void kpack(PtrTab tab, _Float16* __restrict__ w16, float* __restrict__ w32,
                      float* __restrict__ stats) {
  const int t = threadIdx.x, NT = 256;
  for (int e = 0; e < 2; ++e) {
    const float* const* P = tab.p + e * 47;
    _Float16* h = w16 + e * ESTF16;
    float* f = w32 + e * ESTF32;
    copyh_(h + 0,     P[40], 8192, t, NT);   // q_w (64,128)
    copyh_(h + 8192,  P[36], 8192, t, NT);   // k_w
    copyh_(h + 16384, P[46], 8192, t, NT);   // v_w
    copyh_(h + 24576, P[38], 8192, t, NT);   // o_w (128,64)
    copyf_(f + OFF_FC1, P[2], 64, t, NT);
    copyf_(f + OFF_FC2, P[3], 48, t, NT);
    copyf_(f + OFF_FC3, P[4], 64, t, NT);
    copyf_(f + OFF_CONVW, P[1], 32, t, NT);
    copyf_(f + OFF_CONVB, P[0], 8, t, NT);
    copyf_(f + OFF_SE1B, P[41], 4, t, NT);
    copyf_(f + OFF_SE1W, P[42], 32, t, NT);
    copyf_(f + OFF_SE2B, P[43], 8, t, NT);
    copyf_(f + OFF_SE2W, P[44], 32, t, NT);
    copyf_(f + OFF_QB, P[39], 64, t, NT);
    copyf_(f + OFF_KB, P[35], 64, t, NT);
    copyf_(f + OFF_VB, P[45], 64, t, NT);
    copyf_(f + OFF_OB, P[37], 128, t, NT);
    for (int L = 0; L < 3; ++L) {
      const int gb = 5 + 10 * L;
      float* fl = f + OFF_LAYER + L * LSTRIDE;
      copyh_(h + 32768 + 16384 * L, P[gb + 9], 16384, t, NT);  // pj_w (128,128)
      copyf_(fl + LOFF_BNB, P[gb + 0], 32, t, NT);
      copyf_(fl + LOFF_BNG, P[gb + 1], 32, t, NT);
      copyf_(fl + LOFF_F1B, P[gb + 2], 64, t, NT);
      copyf_(fl + LOFF_F1W, P[gb + 3], 512, t, NT);
      copyf_(fl + LOFF_F2B, P[gb + 4], 8, t, NT);
      copyf_(fl + LOFF_F2W, P[gb + 5], 256, t, NT);
      copyf_(fl + LOFF_LNB, P[gb + 6], 128, t, NT);
      copyf_(fl + LOFF_LNG, P[gb + 7], 128, t, NT);
      copyf_(fl + LOFF_PJB, P[gb + 8], 128, t, NT);
    }
  }
  for (int i = t; i < 2 * 3 * 64; i += NT) stats[i] = 0.f;
}

// ---------------------------------------------------------------------------
// K1: attention front end + conv + SE.  One wave per bt.
// ---------------------------------------------------------------------------
__global__ __launch_bounds__(32) void kfront(const float* __restrict__ cin,
                                             const _Float16* __restrict__ w16,
                                             const float* __restrict__ w32,
                                             float* __restrict__ x8out) {
  __shared__ __align__(16) float    p4[4][128];
  __shared__ __align__(16) float    cc3[3][128];
  __shared__ __align__(16) _Float16 q16[16][128];
  __shared__ __align__(16) _Float16 k16[16][128];
  __shared__ __align__(16) _Float16 QtKt[2][64][32];   // transposed Q,K (K-dim padded 16->32)
  __shared__ __align__(16) _Float16 VA[16][64];        // V, A-fragment layout
  __shared__ __align__(16) float    scs[64][64];       // attention scores
  __shared__ __align__(16) _Float16 awb[64][64];       // softmax(aw), B^T layout
  __shared__ __align__(16) _Float16 aA[16][64];        // attn output, A layout
  __shared__ __align__(16) float    a2s[16][128];

  const int bt = blockIdx.x;
  const int lane = threadIdx.x;
  const float* cbt = cin + (size_t)bt * 7 * 128;

  const _Float16* qw = w16;
  const _Float16* kw = w16 + 8192;
  const _Float16* vw = w16 + 16384;
  const _Float16* ow = w16 + 24576;
  __builtin_prefetch(qw);
  __builtin_prefetch(kw);
  __builtin_prefetch(vw);
  __builtin_prefetch(ow);

  for (int i = lane; i < 4 * 128; i += 32) p4[i >> 7][i & 127] = cbt[i];
  for (int i = lane; i < 3 * 128; i += 32) cc3[i >> 7][i & 127] = cbt[3 * 128 + i];
  for (int i = lane; i < 2 * 64 * 32; i += 32) ((_Float16*)QtKt)[i] = (_Float16)0.f;
  __syncthreads();

  // fc1: q = fc1_w(16,4) @ p ; fc2: kin = fc2_w(16,3) @ cc
  const float* fc1 = w32 + OFF_FC1;
  const float* fc2 = w32 + OFF_FC2;
  for (int i = lane; i < 16 * 128; i += 32) {
    const int m = i >> 7, w = i & 127;
    float s = 0.f, s2 = 0.f;
#pragma unroll
    for (int c = 0; c < 4; ++c) s += fc1[m * 4 + c] * p4[c][w];
#pragma unroll
    for (int c = 0; c < 3; ++c) s2 += fc2[m * 3 + c] * cc3[c][w];
    q16[m][w] = (_Float16)s;
    k16[m][w] = (_Float16)s2;
  }
  __syncthreads();

  const float* qb = w32 + OFF_QB;
  const float* kb = w32 + OFF_KB;
  const float* vb = w32 + OFF_VB;
  const float* ob = w32 + OFF_OB;
  const int n = lane & 15, hi = lane >> 4;

  // ---- Q = q @ q_w^T ; K,V = kin @ {k_w,v_w}^T   (M=16, N=64, K=128) -------
  // A fragments are invariant across nt: load once.  B fragments for each nt
  // are preloaded as a batch so the 12 WMMAs issue without per-load waits.
  {
    hvec16 aQ[4], aK[4];
#pragma unroll
    for (int kt = 0; kt < 4; ++kt) {
      aQ[kt] = frag16(&q16[0][0] + kt * 32, 128);
      aK[kt] = frag16(&k16[0][0] + kt * 32, 128);
    }
#pragma unroll
    for (int nt = 0; nt < 4; ++nt) {
      hvec16 bq[4], bk[4], bv[4];
#pragma unroll
      for (int kt = 0; kt < 4; ++kt) {
        bq[kt] = frag16(qw + nt * 16 * 128 + kt * 32, 128);
        bk[kt] = frag16(kw + nt * 16 * 128 + kt * 32, 128);
        bv[kt] = frag16(vw + nt * 16 * 128 + kt * 32, 128);
      }
      fvec8 aq = zero8(), ak = zero8(), av = zero8();
#pragma unroll
      for (int kt = 0; kt < 4; ++kt) {
        aq = WMMA16(aQ[kt], bq[kt], aq);
        ak = WMMA16(aK[kt], bk[kt], ak);
        av = WMMA16(aK[kt], bv[kt], av);
      }
      const int e = nt * 16 + n;
#pragma unroll
      for (int r = 0; r < 8; ++r) {
        const int tok = r + 8 * hi;
        QtKt[0][e][tok] = (_Float16)(aq[r] + qb[e]);
        QtKt[1][e][tok] = (_Float16)(ak[r] + kb[e]);
        VA[tok][e]      = (_Float16)(av[r] + vb[e]);
      }
    }
  }
  __syncthreads();

  // ---- scores[e][f] = sum_n Q[n][e] K[n][f]  (M=N=64, K=16 padded to 32) ---
  {
    hvec16 kfr[4];
#pragma unroll
    for (int nt = 0; nt < 4; ++nt) kfr[nt] = frag16(&QtKt[1][nt * 16][0], 32);
#pragma unroll
    for (int mt = 0; mt < 4; ++mt) {
      hvec16 a = frag16(&QtKt[0][mt * 16][0], 32);
#pragma unroll
      for (int nt = 0; nt < 4; ++nt) {
        fvec8 acc = zero8();
        acc = WMMA16(a, kfr[nt], acc);
#pragma unroll
        for (int r = 0; r < 8; ++r) scs[mt * 16 + r + 8 * hi][nt * 16 + n] = acc[r];
      }
    }
  }
  __syncthreads();

  // softmax over f with scale 16^-0.5 = 0.25
  for (int e = lane; e < 64; e += 32) {
    float mx = -1e30f;
    for (int f = 0; f < 64; ++f) mx = fmaxf(mx, scs[e][f]);
    float sum = 0.f;
    for (int f = 0; f < 64; ++f) { float v = __expf(0.25f * (scs[e][f] - mx)); scs[e][f] = v; sum += v; }
    const float inv = 1.f / sum;
    for (int f = 0; f < 64; ++f) awb[e][f] = (_Float16)(scs[e][f] * inv);
  }
  __syncthreads();

  // ---- a[n][e] = sum_f aw[e][f] V[n][f]   (M=16, N=64, K=64) --------------
  {
    hvec16 va[2];
#pragma unroll
    for (int kt = 0; kt < 2; ++kt) va[kt] = frag16(&VA[0][0] + kt * 32, 64);
#pragma unroll
    for (int nt = 0; nt < 4; ++nt) {
      hvec16 b0 = frag16(&awb[nt * 16][0], 64);
      hvec16 b1 = frag16(&awb[nt * 16][0] + 32, 64);
      fvec8 acc = zero8();
      acc = WMMA16(va[0], b0, acc);
      acc = WMMA16(va[1], b1, acc);
#pragma unroll
      for (int r = 0; r < 8; ++r) aA[r + 8 * hi][nt * 16 + n] = (_Float16)acc[r];
    }
  }
  __syncthreads();

  // ---- a2 = a @ o_w^T + o_b   (M=16, N=128, K=64) -------------------------
  {
    hvec16 aa[2];
#pragma unroll
    for (int kt = 0; kt < 2; ++kt) aa[kt] = frag16(&aA[0][0] + kt * 32, 64);
#pragma unroll
    for (int nt = 0; nt < 8; ++nt) {
      hvec16 b0 = frag16(ow + nt * 16 * 64, 64);
      hvec16 b1 = frag16(ow + nt * 16 * 64 + 32, 64);
      fvec8 acc = zero8();
      acc = WMMA16(aa[0], b0, acc);
      acc = WMMA16(aa[1], b1, acc);
      const int col = nt * 16 + n;
#pragma unroll
      for (int r = 0; r < 8; ++r) a2s[r + 8 * hi][col] = acc[r] + ob[col];
    }
  }
  __syncthreads();

  // x4 = p + fc3 @ a2 ; x8 = conv_w @ x4 + conv_b ; SE
  const float* fc3 = w32 + OFF_FC3;
  const float* cw = w32 + OFF_CONVW;
  const float* cb = w32 + OFF_CONVB;
  const int w0 = lane * 4;
  float x8r[8][4];
#pragma unroll
  for (int j = 0; j < 4; ++j) {
    const int w = w0 + j;
    float xc[4];
#pragma unroll
    for (int c4 = 0; c4 < 4; ++c4) {
      float s = p4[c4][w];
      for (int nn = 0; nn < 16; ++nn) s += fc3[c4 * 16 + nn] * a2s[nn][w];
      xc[c4] = s;
    }
#pragma unroll
    for (int c8 = 0; c8 < 8; ++c8) {
      float s = cb[c8];
#pragma unroll
      for (int c4 = 0; c4 < 4; ++c4) s += cw[c8 * 4 + c4] * xc[c4];
      x8r[c8][j] = s;
    }
  }
  float s8[8];
#pragma unroll
  for (int c8 = 0; c8 < 8; ++c8)
    s8[c8] = wave_sum(x8r[c8][0] + x8r[c8][1] + x8r[c8][2] + x8r[c8][3]) * (1.f / 128.f);
  const float* s1w = w32 + OFF_SE1W; const float* s1b = w32 + OFF_SE1B;
  const float* s2w = w32 + OFF_SE2W; const float* s2b = w32 + OFF_SE2B;
  float t1[4];
#pragma unroll
  for (int j = 0; j < 4; ++j) {
    float s = s1b[j];
#pragma unroll
    for (int c8 = 0; c8 < 8; ++c8) s += s1w[j * 8 + c8] * s8[c8];
    t1[j] = fmaxf(s, 0.f);
  }
  float* xo = x8out + (size_t)bt * 1024;
#pragma unroll
  for (int c8 = 0; c8 < 8; ++c8) {
    float s = s2b[c8];
#pragma unroll
    for (int j = 0; j < 4; ++j) s += s2w[c8 * 4 + j] * t1[j];
    const float gate = sigmoidf_(s);
#pragma unroll
    for (int j = 0; j < 4; ++j) xo[c8 * 128 + w0 + j] = gate * x8r[c8][j];
  }
}

// ---------------------------------------------------------------------------
// gMLP helpers: recompute LN + f1 + gelu per column.
// ---------------------------------------------------------------------------
__global__ __launch_bounds__(32) void kgstats(const float* __restrict__ x8,
                                              const float* __restrict__ wL,
                                              float* __restrict__ stats) {
  const int bt = blockIdx.x, lane = threadIdx.x, w0 = lane * 4;
  const float* xb = x8 + (size_t)bt * 1024;
  float xr[8][4], mean[8], inv[8];
#pragma unroll
  for (int c = 0; c < 8; ++c) {
#pragma unroll
    for (int j = 0; j < 4; ++j) xr[c][j] = xb[c * 128 + w0 + j];
    float s = xr[c][0] + xr[c][1] + xr[c][2] + xr[c][3];
    float q = xr[c][0]*xr[c][0] + xr[c][1]*xr[c][1] + xr[c][2]*xr[c][2] + xr[c][3]*xr[c][3];
    s = wave_sum(s); q = wave_sum(q);
    mean[c] = s * (1.f / 128.f);
    inv[c] = rsqrtf(q * (1.f / 128.f) - mean[c] * mean[c] + 1e-5f);
  }
  const float* lng = wL + LOFF_LNG; const float* lnb = wL + LOFF_LNB;
  const float* f1w = wL + LOFF_F1W; const float* f1b = wL + LOFF_F1B;
  float vs[32], vq[32];
#pragma unroll
  for (int r = 0; r < 32; ++r) { vs[r] = 0.f; vq[r] = 0.f; }
  for (int j = 0; j < 4; ++j) {
    const int w = w0 + j;
    float hc[8];
#pragma unroll
    for (int c = 0; c < 8; ++c) hc[c] = (xr[c][j] - mean[c]) * inv[c] * lng[w] + lnb[w];
#pragma unroll
    for (int r2 = 0; r2 < 32; ++r2) {
      const int r = 32 + r2;
      float t = f1b[r];
#pragma unroll
      for (int c = 0; c < 8; ++c) t += f1w[r * 8 + c] * hc[c];
      const float g = geluf_(t);
      vs[r2] += g; vq[r2] += g * g;
    }
  }
#pragma unroll
  for (int r2 = 0; r2 < 32; ++r2) {
    const float a = wave_sum(vs[r2]);
    const float b = wave_sum(vq[r2]);
    if (lane == 0) { atomicAdd(&stats[r2], a); atomicAdd(&stats[32 + r2], b); }
  }
}

__global__ __launch_bounds__(32) void kgapply(float* __restrict__ x8,
                                              const _Float16* __restrict__ pjw,
                                              const float* __restrict__ wL,
                                              const float* __restrict__ stats) {
  __shared__ __align__(16) float    uu[32][128];
  __shared__ __align__(16) _Float16 vA[32][128];
  __shared__ float sbs[32], sbh[32];

  const int bt = blockIdx.x, lane = threadIdx.x, w0 = lane * 4;
  float* xb = x8 + (size_t)bt * 1024;
  __builtin_prefetch(pjw);

  float xr[8][4], mean[8], inv[8];
#pragma unroll
  for (int c = 0; c < 8; ++c) {
#pragma unroll
    for (int j = 0; j < 4; ++j) xr[c][j] = xb[c * 128 + w0 + j];
    float s = xr[c][0] + xr[c][1] + xr[c][2] + xr[c][3];
    float q = xr[c][0]*xr[c][0] + xr[c][1]*xr[c][1] + xr[c][2]*xr[c][2] + xr[c][3]*xr[c][3];
    s = wave_sum(s); q = wave_sum(q);
    mean[c] = s * (1.f / 128.f);
    inv[c] = rsqrtf(q * (1.f / 128.f) - mean[c] * mean[c] + 1e-5f);
  }
  // fold BN stats (over batch*width) with bn_g/bn_b into scale+shift per channel
  {
    const float* bng = wL + LOFF_BNG; const float* bnb = wL + LOFF_BNB;
    const float Ninv = 1.f / (4096.f * 128.f);
    const float bm = stats[lane] * Ninv;
    const float bv = stats[32 + lane] * Ninv - bm * bm;
    const float sc = rsqrtf(bv + 1e-5f) * bng[lane];
    sbs[lane] = sc;
    sbh[lane] = bnb[lane] - bm * sc;
  }
  __syncthreads();

  const float* lng = wL + LOFF_LNG; const float* lnb = wL + LOFF_LNB;
  const float* f1w = wL + LOFF_F1W; const float* f1b = wL + LOFF_F1B;
  for (int j = 0; j < 4; ++j) {
    const int w = w0 + j;
    float hc[8];
#pragma unroll
    for (int c = 0; c < 8; ++c) hc[c] = (xr[c][j] - mean[c]) * inv[c] * lng[w] + lnb[w];
#pragma unroll
    for (int r = 0; r < 64; ++r) {
      float t = f1b[r];
#pragma unroll
      for (int c = 0; c < 8; ++c) t += f1w[r * 8 + c] * hc[c];
      const float g = geluf_(t);
      if (r < 32) uu[r][w] = g;
      else        vA[r - 32][w] = (_Float16)(g * sbs[r - 32] + sbh[r - 32]);
    }
  }
  __syncthreads();

  // ---- v_proj = v_norm @ pj_w^T + pj_b ; uu <- u * v_proj  (M=32,N=128,K=128)
  // A fragments (8) hoisted; B fragments batched per nt and reused for both mt.
  {
    const float* pjb = wL + LOFF_PJB;
    const int n = lane & 15, hi = lane >> 4;
    hvec16 afr[2][4];
#pragma unroll
    for (int mt = 0; mt < 2; ++mt)
#pragma unroll
      for (int kt = 0; kt < 4; ++kt)
        afr[mt][kt] = frag16(&vA[mt * 16][0] + kt * 32, 128);
#pragma unroll
    for (int nt = 0; nt < 8; ++nt) {
      hvec16 b[4];
#pragma unroll
      for (int kt = 0; kt < 4; ++kt) b[kt] = frag16(pjw + nt * 16 * 128 + kt * 32, 128);
      const int col = nt * 16 + n;
#pragma unroll
      for (int mt = 0; mt < 2; ++mt) {
        fvec8 acc = zero8();
#pragma unroll
        for (int kt = 0; kt < 4; ++kt) acc = WMMA16(afr[mt][kt], b[kt], acc);
#pragma unroll
        for (int r = 0; r < 8; ++r) {
          const int row = mt * 16 + r + 8 * hi;
          uu[row][col] = uu[row][col] * (acc[r] + pjb[col]);
        }
      }
    }
  }
  __syncthreads();

  // x = f2_w @ (u*v) + f2_b + residual
  const float* f2w = wL + LOFF_F2W; const float* f2b = wL + LOFF_F2B;
#pragma unroll
  for (int j = 0; j < 4; ++j) {
    const int w = w0 + j;
#pragma unroll
    for (int c = 0; c < 8; ++c) {
      float s = xr[c][j] + f2b[c];
      for (int r = 0; r < 32; ++r) s += f2w[c * 32 + r] * uu[r][w];
      xb[c * 128 + w] = s;
    }
  }
}

// ---------------------------------------------------------------------------
// K5: act()^2 at width 128, lerp 128 -> 513, emit mag/ang into d_out.
// ---------------------------------------------------------------------------
__global__ __launch_bounds__(128) void kinterp(const float* __restrict__ x8scale,
                                               const float* __restrict__ x8shift,
                                               const float* __restrict__ g,
                                               float* __restrict__ magv,
                                               float* __restrict__ angv) {
  __shared__ float scw[8][128], shw[8][128];
  const int bt = blockIdx.x, tid = threadIdx.x;
  for (int i = tid; i < 1024; i += 128) {
    const float a = softplusf_(x8scale[(size_t)bt * 1024 + i]);
    const float b = sigmoidf_(x8shift[(size_t)bt * 1024 + i]);
    scw[i >> 7][i & 127] = a * a;
    shw[i >> 7][i & 127] = b * b;
  }
  __syncthreads();
  const float gv = g[bt];
  for (int idx = tid; idx < 8 * 513; idx += 128) {
    const int c = idx / 513, f = idx - 513 * c;
    float pos = ((float)f + 0.5f) * (128.f / 513.f) - 0.5f;
    pos = fminf(fmaxf(pos, 0.f), 127.f);
    const int i0 = (int)pos;
    const float w = pos - (float)i0;
    const int i1 = min(i0 + 1, 127);
    const float shv = shw[c][i0] * (1.f - w) + shw[c][i1] * w;
    const float scv = scw[c][i0] * (1.f - w) + scw[c][i1] * w;
    const float ang = gv + (shv * 0.95f + 0.05f);
    const float den = fmaxf(ang, 1.f);
    const size_t o = (size_t)(bt * 8 + c) * 513 + f;
    magv[o] = scv / (den * den);
    angv[o] = ang;
  }
}

// ---------------------------------------------------------------------------
// 1024-pt radix-2 in-LDS FFT (input pre-scrambled bit-reversed; DIT).
// ---------------------------------------------------------------------------
__device__ void fft1024(float2* a, int tid, int nthreads, float sign) {
  for (int s = 1; s <= 10; ++s) {
    const int m = 1 << s, half = m >> 1;
    __syncthreads();
    for (int k = tid; k < 512; k += nthreads) {
      const int grp = k / half, j = k - grp * half;
      const int i0 = grp * m + j, i1 = i0 + half;
      float sn, cs;
      __sincosf(sign * 2.0f * PI_F * (float)j / (float)m, &sn, &cs);
      const float2 u = a[i0], t = a[i1];
      const float tr = t.x * cs - t.y * sn;
      const float ti = t.x * sn + t.y * cs;
      a[i0] = make_float2(u.x + tr, u.y + ti);
      a[i1] = make_float2(u.x - tr, u.y - ti);
    }
  }
  __syncthreads();
}

__global__ __launch_bounds__(256) void kfft_fwd(const float* __restrict__ x,
                                                float* __restrict__ X) {
  __shared__ float2 buf[1024];
  const int bt = blockIdx.x, tid = threadIdx.x;
  for (int i = tid; i < 1024; i += 256)
    buf[__brev((unsigned)i) >> 22] = make_float2(x[(size_t)bt * 1024 + i], 0.f);
  fft1024(buf, tid, 256, -1.f);
  for (int f = tid; f <= 512; f += 256) {
    X[(size_t)bt * 1026 + 2 * f]     = buf[f].x;
    X[(size_t)bt * 1026 + 2 * f + 1] = buf[f].y;
  }
}

// K7: Zsum[f] = sum_c softplus(proj_c)*mag*exp(i*48*ang*omega_f), Y = X*Zsum,
// Hermitian extend, 1024-pt iFFT, write real part -> out.
__global__ __launch_bounds__(256) void kcombine(const float* __restrict__ X,
                                                const float* __restrict__ magv,
                                                const float* __restrict__ angv,
                                                const float* __restrict__ projraw,
                                                float* __restrict__ out) {
  __shared__ float2 Y[513];
  __shared__ float2 buf[1024];
  const int bt = blockIdx.x, tid = threadIdx.x;
  float p8[8];
#pragma unroll
  for (int c = 0; c < 8; ++c) p8[c] = softplusf_(projraw[c]);
  for (int f = tid; f <= 512; f += 256) {
    const float xr = X[(size_t)bt * 1026 + 2 * f];
    const float xi = X[(size_t)bt * 1026 + 2 * f + 1];
    const float om = -PI_F * (float)f / 512.f;
    float zr = 0.f, zi = 0.f;
#pragma unroll
    for (int c = 0; c < 8; ++c) {
      const size_t o = (size_t)(bt * 8 + c) * 513 + f;
      const float m = magv[o];
      float sn, cs;
      __sincosf(48.0f * angv[o] * om, &sn, &cs);
      zr += p8[c] * m * cs;
      zi += p8[c] * m * sn;
    }
    Y[f] = make_float2(xr * zr - xi * zi, xr * zi + xi * zr);
  }
  __syncthreads();
  for (int i = tid; i < 1024; i += 256) {
    const float2 v = (i <= 512) ? Y[i] : make_float2(Y[1024 - i].x, -Y[1024 - i].y);
    buf[__brev((unsigned)i) >> 22] = v;
  }
  fft1024(buf, tid, 256, +1.f);
  for (int t = tid; t < 1024; t += 256)
    out[(size_t)bt * 1024 + t] = buf[t].x * (1.0f / 1024.0f);
}

// ---------------------------------------------------------------------------
extern "C" void kernel_launch(void* const* d_in, const int* in_sizes, int n_in,
                              void* d_out, int out_size, void* d_ws, size_t ws_size,
                              hipStream_t stream) {
  (void)in_sizes; (void)out_size; (void)ws_size;
  const float* x    = (const float*)d_in[0];
  const float* c    = (const float*)d_in[1];
  const float* g    = (const float*)d_in[2];
  const float* proj = (const float*)d_in[3];

  PtrTab tab;
  for (int i = 0; i < 47; ++i) {
    const int is_ = 4 + i, ih_ = 51 + i;
    tab.p[i]      = (const float*)d_in[(is_ < n_in) ? is_ : 0];
    tab.p[47 + i] = (const float*)d_in[(ih_ < n_in) ? ih_ : 0];
  }

  char* ws = (char*)d_ws;
  size_t off = 0;
  auto carve = [&](size_t bytes) -> char* {
    char* p = ws + off;
    off = (off + bytes + 255) & ~(size_t)255;
    return p;
  };
  _Float16* w16 = (_Float16*)carve((size_t)2 * ESTF16 * sizeof(_Float16));
  float* w32    = (float*)carve((size_t)2 * ESTF32 * sizeof(float));
  float* stats  = (float*)carve((size_t)2 * 3 * 64 * sizeof(float));
  float* x8     = (float*)carve((size_t)2 * 4096 * 1024 * sizeof(float));
  float* Xs     = (float*)carve((size_t)4096 * 1026 * sizeof(float));

  float* outp = (float*)d_out;
  float* magv = outp + (size_t)4096 * 1024;
  float* angv = magv + (size_t)4096 * 8 * 513;

  kpack<<<1, 256, 0, stream>>>(tab, w16, w32, stats);
  for (int e = 0; e < 2; ++e) {            // e=0: scale, e=1: shift
    _Float16* w16e = w16 + (size_t)e * ESTF16;
    float* w32e = w32 + (size_t)e * ESTF32;
    float* x8e = x8 + (size_t)e * 4096 * 1024;
    kfront<<<4096, 32, 0, stream>>>(c, w16e, w32e, x8e);
    for (int L = 0; L < 3; ++L) {
      const float* wL = w32e + OFF_LAYER + L * LSTRIDE;
      float* st = stats + (size_t)(e * 3 + L) * 64;
      kgstats<<<4096, 32, 0, stream>>>(x8e, wL, st);
      kgapply<<<4096, 32, 0, stream>>>(x8e, w16e + 32768 + 16384 * L, wL, st);
    }
  }
  kfft_fwd<<<4096, 256, 0, stream>>>(x, Xs);
  kinterp<<<4096, 128, 0, stream>>>(x8, x8 + (size_t)4096 * 1024, g, magv, angv);
  kcombine<<<4096, 256, 0, stream>>>(Xs, magv, angv, proj, outp);
}
